// MambaModel_62612033241721
// MI455X (gfx1250) — compile-verified, matrix operand order checked by reading, not run on hardware
//
#include <hip/hip_runtime.h>

typedef __attribute__((ext_vector_type(16))) __bf16 v16bf;
typedef __attribute__((ext_vector_type(8)))  float  v8f;
typedef __attribute__((ext_vector_type(4)))  int    v4i;

#define B_SZ    4
#define L_SEQ   2048
#define INDIM   256
#define DMODEL  1024
#define DSTATE  16
#define DCONV   4
#define DINNER  2048
#define DTRANK  64
#define NLAYERS 4
#define M_ROWS  (B_SZ * L_SEQ)   // 8192

#define GF_BIAS     1
#define GF_RESID    2
#define GF_SOFTPLUS 4

// gfx1250 async global->LDS copy path (ASYNCcnt), guarded so compile cannot regress
#if defined(__has_builtin)
# if __has_builtin(__builtin_amdgcn_global_load_async_to_lds_b128)
#  define HAVE_ASYNC_LDS 1
# endif
#endif
#ifndef HAVE_ASYNC_LDS
# define HAVE_ASYNC_LDS 0
#endif

#if HAVE_ASYNC_LDS
typedef v4i __attribute__((address_space(1)))* gptr_v4i;   // global
typedef v4i __attribute__((address_space(3)))* lptr_v4i;   // LDS
#endif

__device__ __forceinline__ void wait_async0() {
#if defined(__has_builtin) && __has_builtin(__builtin_amdgcn_s_wait_asynccnt)
    __builtin_amdgcn_s_wait_asynccnt(0);
#else
    asm volatile("s_wait_asynccnt 0x0" ::: "memory");
#endif
}

__device__ __forceinline__ unsigned short f2bf(float f) {
    unsigned int u = __float_as_uint(f);
    u += 0x7FFFu + ((u >> 16) & 1u);   // round-to-nearest-even
    return (unsigned short)(u >> 16);
}
__device__ __forceinline__ float sigm(float x) { return 1.f / (1.f + __expf(-x)); }

// ---------------- casts ----------------
__global__ void k_cast(const float* __restrict__ s, unsigned short* __restrict__ d, int n) {
    int i = blockIdx.x * blockDim.x + threadIdx.x;
    if (i < n) d[i] = f2bf(s[i]);
}
__global__ void k_cast2d(const float* __restrict__ s, int ld, int cols,
                         unsigned short* __restrict__ d, int n) {
    int i = blockIdx.x * blockDim.x + threadIdx.x;
    if (i < n) {
        int r = i / cols, c = i - r * cols;
        d[i] = f2bf(s[(size_t)r * ld + c]);
    }
}

// ---------------- WMMA bf16 GEMM: C = A(MxK) @ B(KxN) [+bias][+resid][softplus] ----------------
// block tile 64(M) x 256(N), 8 waves, wave tile 32x64 = 2x4 WMMA 16x16x32 tiles, K-step 32
__global__ __launch_bounds__(256) void k_gemm_bf16(
    const unsigned short* __restrict__ A, const unsigned short* __restrict__ Bw,
    const float* __restrict__ bias, const float* resid,
    float* C, int M, int N, int K, int flags)
{
    __shared__ unsigned int lds_a[64 * 16];    // 64 rows x 32 bf16 (16 dwords/row)  = 4 KB
    __shared__ unsigned int lds_b[256 * 16];   // 256 cols x 32 bf16, transposed [n][kd] = 16 KB

    const int tid  = threadIdx.x;
    const int lane = tid & 31;
    const int wave = tid >> 5;
    const int waveM = (wave >> 2) * 32;        // 2 wave-rows of 32
    const int waveN = (wave & 3) * 64;         // 4 wave-cols of 64
    const int blockM = blockIdx.y * 64;
    const int blockN = blockIdx.x * 256;
    const int hi = (lane >= 16) ? 1 : 0;

    v8f acc[2][4] = {};

    // ---- per-thread staging pointers (strength-reduced: incremented per K-step) ----
#if HAVE_ASYNC_LDS
    // one async b128 per lane: lane copies 16 bytes of the 64x32 A tile
    const char* aB = (const char*)A + ((size_t)(blockM + (tid >> 2)) * K) * 2 + (size_t)(tid & 3) * 16;
    unsigned int* ldsAdst = lds_a + ((tid >> 2) * 16 + (tid & 3) * 4);
#else
    const char* aB = (const char*)A + ((size_t)(blockM + (tid >> 4)) * K) * 2 + (size_t)(tid & 15) * 4;
#endif
    const int nB  = tid;                       // column within tile (0..255)
    const int gnB = blockN + nB;
    const bool fullN = (blockN + 256) <= N;
    const unsigned short* bp = Bw + gnB;       // column base; += 32*N per K-step

    for (int k0 = 0; k0 < K; k0 += 32) {
        // ---- stage A tile (64x32 bf16) ----
#if HAVE_ASYNC_LDS
        __builtin_amdgcn_global_load_async_to_lds_b128(
            (gptr_v4i)aB, (lptr_v4i)ldsAdst, 0, 0);
#else
        #pragma unroll
        for (int i = 0; i < 4; ++i)
            lds_a[(tid >> 4) * 16 + (tid & 15) + i * 256] =
                *(const unsigned int*)(aB + (size_t)i * 32 * K);   // 16 rows = 32K bytes
#endif
        aB += 64;

        // ---- stage B tile (32x256 bf16), transposed so K pairs are dword-contiguous ----
        if (fullN || gnB < N) {
            const unsigned short* p = bp;
            #pragma unroll
            for (int i = 0; i < 16; ++i) {
                unsigned int b0 = p[0];
                unsigned int b1 = p[N];
                lds_b[nB * 16 + i] = (b1 << 16) | b0;
                p += 2 * (size_t)N;
            }
        } else {
            #pragma unroll
            for (int i = 0; i < 16; ++i) lds_b[nB * 16 + i] = 0;
        }
        bp += 32 * (size_t)N;

        // prefetch next tiles while this one is consumed
        if (k0 + 32 < K) {
            __builtin_prefetch(aB, 0, 0);
            __builtin_prefetch(bp, 0, 0);
        }

#if HAVE_ASYNC_LDS
        wait_async0();
#endif
        __syncthreads();

        // ---- fragments per documented bf16 VGPR layout:
        // lanes 0-15: K 0-7 then 16-23 ; lanes 16-31: K 8-15 then 24-31
        union Frag { uint4 q[2]; v16bf v; };
        Frag fa[2], fb[4];
        #pragma unroll
        for (int mi = 0; mi < 2; ++mi) {
            int base = (waveM + mi * 16 + (lane & 15)) * 16 + hi * 4;
            fa[mi].q[0] = *(const uint4*)&lds_a[base];
            fa[mi].q[1] = *(const uint4*)&lds_a[base + 8];
        }
        #pragma unroll
        for (int ni = 0; ni < 4; ++ni) {
            int base = (waveN + ni * 16 + (lane & 15)) * 16 + hi * 4;
            fb[ni].q[0] = *(const uint4*)&lds_b[base];
            fb[ni].q[1] = *(const uint4*)&lds_b[base + 8];
        }
        #pragma unroll
        for (int mi = 0; mi < 2; ++mi)
            #pragma unroll
            for (int ni = 0; ni < 4; ++ni)
                acc[mi][ni] = __builtin_amdgcn_wmma_f32_16x16x32_bf16(
                    false, fa[mi].v, false, fb[ni].v, (short)0, acc[mi][ni], false, false);
        __syncthreads();
    }

    // ---- epilogue: C/D layout -> row = e + 8*hi, col = lane&15 ----
    #pragma unroll
    for (int mi = 0; mi < 2; ++mi) {
        #pragma unroll
        for (int ni = 0; ni < 4; ++ni) {
            int col = blockN + waveN + ni * 16 + (lane & 15);
            if (col >= N) continue;
            #pragma unroll
            for (int e = 0; e < 8; ++e) {
                int rowG = blockM + waveM + mi * 16 + e + hi * 8;
                float v = acc[mi][ni][e];
                if (flags & GF_BIAS) v += bias[col];
                size_t idx = (size_t)rowG * N + col;
                if (flags & GF_RESID) v += resid[idx];
                if (flags & GF_SOFTPLUS) v = (v > 20.f) ? v : log1pf(__expf(v));
                C[idx] = v;
            }
        }
    }
}

// ---------------- causal depthwise conv (width 4) + SiLU; writes f32 u and bf16 copy ----------------
__global__ void k_conv_silu(const float* __restrict__ xz, const float* __restrict__ cw,
                            const float* __restrict__ cb, float* __restrict__ u,
                            unsigned short* __restrict__ ubf) {
    int idx = blockIdx.x * 256 + threadIdx.x;       // over M_ROWS*DINNER
    int d   = idx & (DINNER - 1);
    int row = idx >> 11;
    int l   = row & (L_SEQ - 1);
    int bb  = row >> 11;
    float acc = cb[d];
    #pragma unroll
    for (int j = 0; j < DCONV; ++j) {
        int ll = l - (DCONV - 1) + j;
        if (ll >= 0)
            acc += xz[((size_t)(bb * L_SEQ + ll) * (2 * DINNER)) + d] * cw[d * DCONV + j];
    }
    float v = acc * sigm(acc);
    u[idx]   = v;
    ubf[idx] = f2bf(v);
}

// ---------------- selective scan: 1 channel/lane, 16 states in regs, B/C chunked in LDS ----------------
__global__ __launch_bounds__(128) void k_scan(
    const float* __restrict__ u, const float* delta_in,
    const float* __restrict__ dbl, const float* __restrict__ A_log,
    const float* __restrict__ Dp, float* y_out)   // y_out aliases delta_in (read-before-write)
{
    __shared__ float bc[64][32];                  // [tt][0..15]=B, [16..31]=C
    int bb = blockIdx.y;
    int d  = blockIdx.x * 128 + threadIdx.x;
    float as[DSTATE], hs[DSTATE];
    #pragma unroll
    for (int s = 0; s < DSTATE; ++s) {
        as[s] = -__expf(A_log[(size_t)d * DSTATE + s]);
        hs[s] = 0.f;
    }
    float dpar = Dp[d];
    for (int t0 = 0; t0 < L_SEQ; t0 += 64) {
        __syncthreads();
        #pragma unroll
        for (int i = 0; i < 16; ++i) {
            int linear = threadIdx.x + i * 128;
            int tt = linear >> 5;
            int j  = linear & 31;
            bc[tt][j] = dbl[((size_t)(bb * L_SEQ + t0 + tt)) * 96 + 64 + j];
        }
        __syncthreads();
        for (int tt = 0; tt < 64; ++tt) {
            size_t off = ((size_t)(bb * L_SEQ + t0 + tt)) * DINNER + d;
            float dl = delta_in[off];
            float ut = u[off];
            float du = dl * ut;
            float y = 0.f;
            #pragma unroll
            for (int s = 0; s < DSTATE; ++s) {
                float dA = __expf(dl * as[s]);
                hs[s] = fmaf(hs[s], dA, du * bc[tt][s]);
                y = fmaf(hs[s], bc[tt][16 + s], y);
            }
            y_out[off] = y + ut * dpar;
        }
    }
}

// ---------------- gate: y *= silu(z); bf16 out for final GEMM ----------------
__global__ void k_gate(const float* __restrict__ y, const float* __restrict__ xz,
                       unsigned short* __restrict__ ybf) {
    int idx = blockIdx.x * 256 + threadIdx.x;
    int d   = idx & (DINNER - 1);
    int row = idx >> 11;
    float z = xz[(size_t)row * (2 * DINNER) + DINNER + d];
    float v = y[idx] * (z * sigm(z));
    ybf[idx] = f2bf(v);
}

// ---------------- final: LayerNorm on last-timestep rows + fc (OUT_DIM=1) ----------------
__global__ __launch_bounds__(256) void k_ln_fc(
    const float* __restrict__ h, const float* __restrict__ g, const float* __restrict__ b2,
    const float* __restrict__ fcw, const float* __restrict__ fcb, float* __restrict__ out) {
    __shared__ float red[256];
    int b = blockIdx.x, tid = threadIdx.x;
    const float* row = h + ((size_t)(b * L_SEQ + (L_SEQ - 1))) * DMODEL;
    float s = 0.f, s2 = 0.f;
    for (int c = tid; c < DMODEL; c += 256) { float v = row[c]; s += v; s2 += v * v; }
    red[tid] = s; __syncthreads();
    for (int o = 128; o > 0; o >>= 1) { if (tid < o) red[tid] += red[tid + o]; __syncthreads(); }
    float mu = red[0] / (float)DMODEL; __syncthreads();
    red[tid] = s2; __syncthreads();
    for (int o = 128; o > 0; o >>= 1) { if (tid < o) red[tid] += red[tid + o]; __syncthreads(); }
    float var = red[0] / (float)DMODEL - mu * mu; __syncthreads();
    float rstd = rsqrtf(var + 1e-5f);
    float acc = 0.f;
    for (int c = tid; c < DMODEL; c += 256) {
        float v = (row[c] - mu) * rstd * g[c] + b2[c];
        acc += v * fcw[c];
    }
    red[tid] = acc; __syncthreads();
    for (int o = 128; o > 0; o >>= 1) { if (tid < o) red[tid] += red[tid + o]; __syncthreads(); }
    if (tid == 0) out[b] = red[0] + fcb[0];
}

extern "C" void kernel_launch(void* const* d_in, const int* in_sizes, int n_in,
                              void* d_out, int out_size, void* d_ws, size_t ws_size,
                              hipStream_t stream) {
    const float* x        = (const float*)d_in[0];
    const float* in_w     = (const float*)d_in[1];
    const float* in_b     = (const float*)d_in[2];
    const float* blk_in_w = (const float*)d_in[3];
    const float* blk_in_b = (const float*)d_in[4];
    const float* conv_w   = (const float*)d_in[5];
    const float* conv_b   = (const float*)d_in[6];
    const float* xproj_w  = (const float*)d_in[7];
    const float* dt_w     = (const float*)d_in[8];
    const float* dt_b     = (const float*)d_in[9];
    const float* A_log    = (const float*)d_in[10];
    const float* Dparam   = (const float*)d_in[11];
    const float* out_w    = (const float*)d_in[12];
    const float* out_b    = (const float*)d_in[13];
    const float* ln_g     = (const float*)d_in[14];
    const float* ln_b     = (const float*)d_in[15];
    const float* fc_w     = (const float*)d_in[16];
    const float* fc_b     = (const float*)d_in[17];

    char* ws = (char*)d_ws;
    float* h     = (float*)(ws);                                   // 32 MB
    float* xz    = (float*)(ws + (32ull  << 20));                  // 128 MB
    float* u     = (float*)(ws + (160ull << 20));                  // 64 MB
    float* delta = (float*)(ws + (224ull << 20));                  // 64 MB (reused as y)
    float* dbl   = (float*)(ws + (288ull << 20));                  // 3 MB
    unsigned short* abuf = (unsigned short*)(ws + (292ull << 20)); // 32 MB bf16 activations
    unsigned short* wbuf = (unsigned short*)(ws + (324ull << 20)); // 8 MB  bf16 weights
    unsigned short* dtbf = (unsigned short*)(ws + (332ull << 20)); // 1 MB

    auto cast = [&](const float* s, unsigned short* d, int n) {
        k_cast<<<(n + 255) / 256, 256, 0, stream>>>(s, d, n);
    };
    auto gemm = [&](const unsigned short* A, const unsigned short* Bw, const float* bias,
                    const float* resid, float* C, int M, int N, int K, int flags) {
        dim3 g((N + 255) / 256, M / 64);
        k_gemm_bf16<<<g, 256, 0, stream>>>(A, Bw, bias, resid, C, M, N, K, flags);
    };

    // input projection: h = x @ in_w + in_b
    cast(x, abuf, M_ROWS * INDIM);
    cast(in_w, wbuf, INDIM * DMODEL);
    gemm(abuf, wbuf, in_b, nullptr, h, M_ROWS, DMODEL, INDIM, GF_BIAS);

    for (int i = 0; i < NLAYERS; ++i) {
        // xz = h @ blk_in_w[i] + blk_in_b[i]
        cast(h, abuf, M_ROWS * DMODEL);
        cast(blk_in_w + (size_t)i * DMODEL * 2 * DINNER, wbuf, DMODEL * 2 * DINNER);
        gemm(abuf, wbuf, blk_in_b + i * 2 * DINNER, nullptr, xz, M_ROWS, 2 * DINNER, DMODEL, GF_BIAS);
        // x = silu(causal_dwconv(x)) -> u (f32) + abuf (bf16)
        k_conv_silu<<<M_ROWS * DINNER / 256, 256, 0, stream>>>(
            xz, conv_w + (size_t)i * DINNER * DCONV, conv_b + i * DINNER, u, abuf);
        // dbl = x @ xproj_w[i]  (dt | B | C)
        cast(xproj_w + (size_t)i * DINNER * 96, wbuf, DINNER * 96);
        gemm(abuf, wbuf, nullptr, nullptr, dbl, M_ROWS, 96, DINNER, 0);
        // delta = softplus(dt @ dt_w[i] + dt_b[i])
        k_cast2d<<<(M_ROWS * DTRANK + 255) / 256, 256, 0, stream>>>(dbl, 96, DTRANK, dtbf, M_ROWS * DTRANK);
        cast(dt_w + (size_t)i * DTRANK * DINNER, wbuf, DTRANK * DINNER);
        gemm(dtbf, wbuf, dt_b + i * DINNER, nullptr, delta, M_ROWS, DINNER, DTRANK, GF_BIAS | GF_SOFTPLUS);
        // selective scan (y written in-place over delta)
        k_scan<<<dim3(DINNER / 128, B_SZ), 128, 0, stream>>>(
            u, delta, dbl, A_log + (size_t)i * DINNER * DSTATE, Dparam + i * DINNER, delta);
        // y *= silu(z) -> bf16 in abuf
        k_gate<<<M_ROWS * DINNER / 256, 256, 0, stream>>>(delta, xz, abuf);
        // h = h + y @ out_w[i] + out_b[i]
        cast(out_w + (size_t)i * DINNER * DMODEL, wbuf, DINNER * DMODEL);
        gemm(abuf, wbuf, out_b + i * DMODEL, h, h, M_ROWS, DMODEL, DINNER, GF_BIAS | GF_RESID);
    }

    k_ln_fc<<<B_SZ, 256, 0, stream>>>(h, ln_g, ln_b, fc_w, fc_b, (float*)d_out);
}
